// ParamEmbeddingGenerator_38070590111960
// MI455X (gfx1250) — compile-verified, matrix operand order checked by reading, not run on hardware
//
#include <hip/hip_runtime.h>

// ---------------------------------------------------------------------------
// Problem constants (match reference)
// ---------------------------------------------------------------------------
constexpr int B  = 8;
constexpr int L  = 4096;
constexpr int D  = 768;
constexpr int NC = 4096;          // compact slots
constexpr int M  = B * NC;        // GEMM rows = 32768

typedef __attribute__((ext_vector_type(16))) __bf16 v16bf;
typedef __attribute__((ext_vector_type(8)))  __bf16 v8bf;
typedef __attribute__((ext_vector_type(8)))  float  v8f;
typedef __attribute__((ext_vector_type(4)))  int    v4i;

#define AS1 __attribute__((address_space(1)))
#define AS3 __attribute__((address_space(3)))

typedef AS1 v4i* gv4i_p;   // pointer to int4 in global (AS1)
typedef AS3 v4i* lv4i_p;   // pointer to int4 in LDS (AS3)

#if __has_builtin(__builtin_amdgcn_global_load_async_to_lds_b128)
#define HAVE_ASYNC_LDS 1
#else
#define HAVE_ASYNC_LDS 0
#endif

__device__ __forceinline__ float fast_tanh(float x) {
#if __has_builtin(__builtin_amdgcn_tanhf)
    return __builtin_amdgcn_tanhf(x);        // V_TANH_F32 trans op
#else
    return tanhf(x);
#endif
}

__device__ __forceinline__ void wait_async0() {
#if __has_builtin(__builtin_amdgcn_s_wait_asynccnt)
    __builtin_amdgcn_s_wait_asynccnt(0);
#else
    asm volatile("s_wait_asynccnt 0x0" ::: "memory");
#endif
}

// ---------------------------------------------------------------------------
// 1) prep: W (DxD f32, k-major) -> Wt (n-major bf16), zero atomic accumulators
// ---------------------------------------------------------------------------
__global__ void prep_kernel(const float* __restrict__ W, __bf16* __restrict__ Wt,
                            int* __restrict__ accums) {
    int idx = blockIdx.x * blockDim.x + threadIdx.x;
    if (idx == 0) { accums[0] = 0; accums[1] = 0; }   // num, den
    if (idx < D * D) {
        int k = idx / D, n = idx % D;
        Wt[n * D + k] = (__bf16)W[idx];
    }
}

// ---------------------------------------------------------------------------
// 2) seg_bounds: lb[b][n] = lower_bound(seg_ids[b], n) for n in [0, NC]
// ---------------------------------------------------------------------------
__global__ void seg_bounds_kernel(const int* __restrict__ seg, int* __restrict__ lb) {
    int idx = blockIdx.x * blockDim.x + threadIdx.x;
    if (idx >= B * (NC + 1)) return;
    int b = idx / (NC + 1), n = idx % (NC + 1);
    const int* s = seg + b * L;
    int lo = 0, hi = L;
    while (lo < hi) { int mid = (lo + hi) >> 1; if (s[mid] < n) lo = mid + 1; else hi = mid; }
    lb[idx] = lo;
}

// ---------------------------------------------------------------------------
// 3) chunk_stats: one block per (b,n) chunk. 256 threads * 3 cols = D.
// ---------------------------------------------------------------------------
__global__ __launch_bounds__(256)
void chunk_stats_kernel(const float* __restrict__ x,     // [B,L,D]
                        const int*   __restrict__ lb,    // [B,NC+1]
                        const int*   __restrict__ pad,   // [B,L]
                        const int*   __restrict__ reg,   // [B,L]
                        const int*   __restrict__ sp,    // [B,L]
                        __bf16*      __restrict__ meanBF,// [M,D]
                        int*         __restrict__ counts,// [M]
                        float* __restrict__ mpad, float* __restrict__ mreg,
                        float* __restrict__ msp, int* __restrict__ denom) {
    const int m = blockIdx.x;
    const int b = m >> 12;            // m / NC
    const int n = m & (NC - 1);       // m % NC
    const int start = lb[b * (NC + 1) + n];
    const int end   = lb[b * (NC + 1) + n + 1];
    const int cnt   = end - start;
    const int col   = threadIdx.x * 3;

    float s0 = 0.f, s1 = 0.f, s2 = 0.f;
    for (int l = start; l < end; ++l) {
        const float* p = x + ((size_t)b * L + l) * D + col;
        s0 += p[0]; s1 += p[1]; s2 += p[2];
    }
    const float inv = 1.0f / (float)(cnt > 1 ? cnt : 1);
    __bf16* o = meanBF + (size_t)m * D + col;
    o[0] = (__bf16)(s0 * inv); o[1] = (__bf16)(s1 * inv); o[2] = (__bf16)(s2 * inv);

    if (threadIdx.x == 0) {
        int ps = 0, rs = 0, ss = 0;
        const int base = b * L;
        for (int l = start; l < end; ++l) { ps += pad[base + l]; rs += reg[base + l]; ss += sp[base + l]; }
        counts[m] = cnt;
        const int mp = (ps > 0) ? 1 : 0;
        mpad[m] = (float)mp;
        mreg[m] = (rs > 0) ? 1.0f : 0.0f;
        const int sa = (cnt > 0 && ss == cnt) ? 1 : 0;      // all() over chunk
        msp[m] = mp ? (float)sa : -1.0f;                    // where(pad==0, -1, sp_all)
        if (rs) atomicAdd(denom, rs);                       // sum(regular_tokens_mask)
    }
}

// ---------------------------------------------------------------------------
// 4) keep[n] = any_b(mask_padding[b][n])
// ---------------------------------------------------------------------------
__global__ void keep_kernel(const float* __restrict__ mpad, int* __restrict__ keep) {
    int n = blockIdx.x * blockDim.x + threadIdx.x;
    if (n >= NC) return;
    int k = 0;
#pragma unroll
    for (int b = 0; b < B; ++b) k |= (mpad[b * NC + n] != 0.0f);
    keep[n] = k;
}

// ---------------------------------------------------------------------------
// 5) finalize masks (x keep), numerator atomic, per-row GEMM gate
// ---------------------------------------------------------------------------
__global__ void finalize_kernel(float* __restrict__ mpad, float* __restrict__ mreg,
                                float* __restrict__ msp, const int* __restrict__ keep,
                                const int* __restrict__ counts,
                                float* __restrict__ gate, int* __restrict__ num) {
    int m = blockIdx.x * blockDim.x + threadIdx.x;
    if (m >= M) return;
    const int n = m & (NC - 1);
    const int k = keep[n];
    const float kf = (float)k;
    mpad[m] *= kf;
    const float r = mreg[m] * kf; mreg[m] = r;
    msp[m] *= kf;
    if (r != 0.0f) atomicAdd(num, 1);                       // sum(mask_regular_o)
    gate[m] = (counts[m] > 0 && k) ? 1.0f : 0.0f;           // counts>0 AND keep
}

// ---------------------------------------------------------------------------
// 6) compression rate scalar
// ---------------------------------------------------------------------------
__global__ void rate_kernel(const int* __restrict__ accums, float* __restrict__ out) {
    out[0] = (float)accums[0] / (float)accums[1];
}

// ---------------------------------------------------------------------------
// 7) WMMA GEMM: out[m][n] = tanh(sum_k meanBF[m][k]*W[k][n] + b[n]) * gate[m]
//    Block tile 128x128, BK=32. 8 waves: 4 (M) x 2 (N); each wave 32x64 =
//    2x4 tiles of v_wmma_f32_16x16x32_bf16.
//    Fast path: tiles whose 128 rows are all gated off (n >= N_CH or empty
//    chunks -> count==0) skip the WMMA loop entirely and stream zeros.
// ---------------------------------------------------------------------------
__global__ __launch_bounds__(256)
void gemm_tanh_kernel(const __bf16* __restrict__ A,    // [M,D] mean, bf16
                      const __bf16* __restrict__ Bt,   // [D,D] W transposed (n-major)
                      const float*  __restrict__ bias, // [D]
                      const float*  __restrict__ gate, // [M]
                      float*        __restrict__ out)  // [M,D]
{
    constexpr int LDT = 40;                       // padded LDS stride (elements)
    __shared__ __align__(16) __bf16 lsA[128 * LDT];
    __shared__ __align__(16) __bf16 lsB[128 * LDT];
    __shared__ int anyGate;

    const int tid = threadIdx.x;
    const int M0  = blockIdx.x * 128;
    const int N0  = blockIdx.y * 128;

    // ---- all-zero tile fast path (>= half of all tiles) ----
    if (tid == 0) anyGate = 0;
    __syncthreads();
    if (tid < 128 && gate[M0 + tid] != 0.0f) anyGate = 1;   // benign race
    __syncthreads();
    if (!anyGate) {
#pragma unroll
        for (int t = 0; t < 16; ++t) {
            const int c   = tid + t * 256;        // 4096 float4 slots = 128x128
            const int row = c >> 5;
            const int q   = (c & 31) * 4;
            *(float4*)&out[(size_t)(M0 + row) * D + N0 + q] = make_float4(0.f, 0.f, 0.f, 0.f);
        }
        return;
    }

    const int lane = tid & 31;
    const int wid  = tid >> 5;
    const int mW   = (wid & 3) * 32;              // wave M offset in tile
    const int nW   = (wid >> 2) * 64;             // wave N offset in tile
    const int lrow = lane & 15;
    const int ksel = (lane < 16) ? 0 : 8;         // K split across half-waves

    v8f acc[2][4];
#pragma unroll
    for (int mt = 0; mt < 2; ++mt)
#pragma unroll
        for (int nt = 0; nt < 4; ++nt)
            acc[mt][nt] = (v8f){0.f, 0.f, 0.f, 0.f, 0.f, 0.f, 0.f, 0.f};

    for (int kb = 0; kb < D; kb += 32) {
        // cooperative tile load: 128 rows x 64B = 512 x 16B chunks, 2 per thread
#pragma unroll
        for (int t = 0; t < 2; ++t) {
            const int c   = tid + t * 256;
            const int row = c >> 2;
            const int kq  = (c & 3) * 8;
#if HAVE_ASYNC_LDS
            __builtin_amdgcn_global_load_async_to_lds_b128(
                (gv4i_p)(uintptr_t)&A[(size_t)(M0 + row) * D + kb + kq],
                (lv4i_p)&lsA[row * LDT + kq], 0, 0);
            __builtin_amdgcn_global_load_async_to_lds_b128(
                (gv4i_p)(uintptr_t)&Bt[(size_t)(N0 + row) * D + kb + kq],
                (lv4i_p)&lsB[row * LDT + kq], 0, 0);
#else
            *(uint4*)&lsA[row * LDT + kq] = *(const uint4*)&A [(size_t)(M0 + row) * D + kb + kq];
            *(uint4*)&lsB[row * LDT + kq] = *(const uint4*)&Bt[(size_t)(N0 + row) * D + kb + kq];
#endif
        }
        if (kb + 32 < D) {  // prefetch next K tile -> global_prefetch_b8
            const int row = tid >> 2, kq = (tid & 3) * 8;
            __builtin_prefetch(&A [(size_t)(M0 + row) * D + kb + 32 + kq], 0, 1);
            __builtin_prefetch(&Bt[(size_t)(N0 + row) * D + kb + 32 + kq], 0, 1);
        }
#if HAVE_ASYNC_LDS
        wait_async0();
#endif
        __syncthreads();

        v16bf af[2], bfr[4];
#pragma unroll
        for (int mt = 0; mt < 2; ++mt) {
            const __bf16* p = &lsA[(mW + mt * 16 + lrow) * LDT + ksel];
            v8bf lo = *(const v8bf*)p;            // K ksel..ksel+7
            v8bf hi = *(const v8bf*)(p + 16);     // K ksel+16..ksel+23
#pragma unroll
            for (int i = 0; i < 8; ++i) { af[mt][i] = lo[i]; af[mt][i + 8] = hi[i]; }
        }
#pragma unroll
        for (int nt = 0; nt < 4; ++nt) {
            const __bf16* p = &lsB[(nW + nt * 16 + lrow) * LDT + ksel];
            v8bf lo = *(const v8bf*)p;
            v8bf hi = *(const v8bf*)(p + 16);
#pragma unroll
            for (int i = 0; i < 8; ++i) { bfr[nt][i] = lo[i]; bfr[nt][i + 8] = hi[i]; }
        }
#pragma unroll
        for (int mt = 0; mt < 2; ++mt)
#pragma unroll
            for (int nt = 0; nt < 4; ++nt)
                acc[mt][nt] = __builtin_amdgcn_wmma_f32_16x16x32_bf16(
                    false, af[mt], false, bfr[nt], (short)0, acc[mt][nt], false, false);
        __syncthreads();
    }

    // fused epilogue: tanh(acc + bias) * gate  (V_TANH_F32 when available)
    const int rAdd = (lane < 16) ? 0 : 8;
#pragma unroll
    for (int mt = 0; mt < 2; ++mt) {
        float g[8];
#pragma unroll
        for (int r = 0; r < 8; ++r) g[r] = gate[M0 + mW + mt * 16 + r + rAdd];
#pragma unroll
        for (int nt = 0; nt < 4; ++nt) {
            const int col = N0 + nW + nt * 16 + lrow;
            const float bc = bias[col];
#pragma unroll
            for (int r = 0; r < 8; ++r) {
                const int row = M0 + mW + mt * 16 + r + rAdd;
                out[(size_t)row * D + col] = fast_tanh(acc[mt][nt][r] + bc) * g[r];
            }
        }
    }
}

// ---------------------------------------------------------------------------
// launch
// ---------------------------------------------------------------------------
extern "C" void kernel_launch(void* const* d_in, const int* in_sizes, int n_in,
                              void* d_out, int out_size, void* d_ws, size_t ws_size,
                              hipStream_t stream) {
    const float* x    = (const float*)d_in[0];   // tensors_batch [B,L,D]
    const int*   seg  = (const int*)  d_in[1];   // seg_ids       [B,L]
    const int*   pad  = (const int*)  d_in[2];   // padding_mask  [B,L]
    const int*   reg  = (const int*)  d_in[3];   // regular_mask  [B,L]
    const int*   sp   = (const int*)  d_in[4];   // seq_pair_mask [B,L]
    const float* W    = (const float*)d_in[5];   // [D,D]
    const float* bias = (const float*)d_in[6];   // [D]

    float* out_compact = (float*)d_out;          // [B,NC,D]
    float* out_mpad    = out_compact + (size_t)M * D;
    float* out_mreg    = out_mpad + M;
    float* out_msp     = out_mreg + M;
    float* out_rate    = out_msp + M;            // scalar

    // workspace carve (~52 MB total)
    size_t off = 0;
    auto carve = [&](size_t bytes) -> void* {
        void* p = (char*)d_ws + off;
        off = (off + bytes + 255) & ~(size_t)255;
        return p;
    };
    int*    lb     = (int*)   carve((size_t)B * (NC + 1) * 4);
    int*    counts = (int*)   carve((size_t)M * 4);
    int*    keep   = (int*)   carve((size_t)NC * 4);
    float*  gate   = (float*) carve((size_t)M * 4);
    int*    accums = (int*)   carve(256);                    // [num, den]
    __bf16* Wt     = (__bf16*)carve((size_t)D * D * 2);
    __bf16* meanBF = (__bf16*)carve((size_t)M * D * 2);

    prep_kernel<<<(D * D + 255) / 256, 256, 0, stream>>>(W, Wt, accums);
    seg_bounds_kernel<<<(B * (NC + 1) + 255) / 256, 256, 0, stream>>>(seg, lb);
    chunk_stats_kernel<<<M, 256, 0, stream>>>(x, lb, pad, reg, sp, meanBF, counts,
                                              out_mpad, out_mreg, out_msp, accums + 1);
    keep_kernel<<<(NC + 255) / 256, 256, 0, stream>>>(out_mpad, keep);
    finalize_kernel<<<(M + 255) / 256, 256, 0, stream>>>(out_mpad, out_mreg, out_msp,
                                                         keep, counts, gate, accums);
    rate_kernel<<<1, 1, 0, stream>>>(accums, out_rate);
    gemm_tanh_kernel<<<dim3(M / 128, D / 128), 256, 0, stream>>>(meanBF, Wt, bias, gate,
                                                                 out_compact);
}
